// EnhancedTFNLayer_80676665688791
// MI455X (gfx1250) — matrix operand order repaired and checked by reading, not compiled, not run on hardware
//
#include <hip/hip_runtime.h>

// ---------------------------------------------------------------------------
// EnhancedTFNLayer for MI455X (gfx1250, wave32, WMMA + TDM)
// B=16, N=2048, D=512, M=512, H=8, DH=64, 3 diffusion steps, dt=0.01
// GEMMs: bf16 x bf16 -> f32 via v_wmma_f32_16x16x32_bf16, operands staged
// into LDS by the Tensor Data Mover (tensor_load_to_lds + s_wait_tensorcnt).
// ---------------------------------------------------------------------------

#define BB 16
#define NN 2048
#define DD 512
#define MM 512
#define HH 8
#define DHD 64
#define PDE_STEPS 3
#define DT_STEP 0.01f

// GEMM tiling: 64x64 C tile per 128-thread block (4 waves, 16x64 per wave),
// K consumed in 64-wide stages. LDS tile rows padded 64->72 elements
// (TDM pad: 32-dword interval, 4-dword amount) for conflict-free ds reads.
#define KCHUNK 64
#define TPITCH 72
#define TILE_BYTES (64 * TPITCH * 2)   // 9216 B per tile

typedef __attribute__((ext_vector_type(8)))  __bf16 v8bf;
typedef __attribute__((ext_vector_type(16))) __bf16 v16bf;
typedef __attribute__((ext_vector_type(8)))  float  v8f;
typedef __attribute__((ext_vector_type(4)))  unsigned int u32x4;
typedef __attribute__((ext_vector_type(8)))  int i32x8;
typedef __attribute__((ext_vector_type(4)))  int i32x4;

#if __has_builtin(__builtin_amdgcn_tensor_load_to_lds) && __has_builtin(__builtin_amdgcn_s_wait_tensorcnt)
#define USE_TDM 1
#else
#define USE_TDM 0
#endif

// ----------------------------- device helpers ------------------------------

__device__ inline float gelu_tanh(float x) {
    const float k0 = 0.7978845608028654f;  // sqrt(2/pi)
    float x3 = x * x * x;
    return 0.5f * x * (1.0f + tanhf(k0 * (x + 0.044715f * x3)));
}

// A fragment: 16 contiguous K per lane (lanes 0-15 K=0..15, 16-31 K=16..31).
__device__ inline v16bf load_a_frag(const __bf16* __restrict__ p) {
    v8bf lo = *(const v8bf*)(p);
    v8bf hi = *(const v8bf*)(p + 8);
    v16bf r;
#pragma unroll
    for (int i = 0; i < 8; ++i) { r[i] = lo[i]; r[i + 8] = hi[i]; }
    return r;
}

// B fragment: 16 contiguous K per lane (lanes 0-15 K=0..15, 16-31 K=16..31).
__device__ inline v16bf load_b_frag(const __bf16* __restrict__ p) {
    v8bf lo = *(const v8bf*)(p);
    v8bf hi = *(const v8bf*)(p + 8);
    v16bf r;
#pragma unroll
    for (int i = 0; i < 8; ++i) { r[i] = lo[i]; r[i + 8] = hi[i]; }
    return r;
}

#if USE_TDM
// Issue a TDM 2-D tile load: [64 rows x 64 k] bf16, row stride `strideElems`,
// into LDS at byte offset lds_off with +8-element row padding.
// D# layout per CDNA5 ISA. 6-arg builtin form:
// (g0, g1, g2, g3, g4, cpol).
__device__ inline void tdm_load_tile(unsigned lds_off, const __bf16* gptr,
                                     unsigned kExtent, unsigned rowExtent,
                                     unsigned long strideElems) {
    unsigned long ga = (unsigned long)gptr;
    u32x4 g0;
    g0[0] = 1u;                                   // count=1 (valid descriptor)
    g0[1] = lds_off;                              // lds_addr (bytes)
    g0[2] = (unsigned)ga;                         // global_addr[31:0]
    g0[3] = (unsigned)((ga >> 32) & 0x01FFFFFFu)  // global_addr[56:32]
            | (2u << 30);                         // type=2 ("image")
    i32x8 g1;
    g1[0] = (int)((1u << 16)      // data_size = 2 bytes
                  | (1u << 20)    // pad_enable
                  | (4u << 22)    // pad_interval: 32 dwords (=64 bf16 row)
                  | (3u << 25));  // pad_amount: 4 dwords (=8 bf16)
    g1[1] = (int)((kExtent & 0xFFFFu) << 16);               // tensor_dim0 lo
    g1[2] = (int)((kExtent >> 16) | ((rowExtent & 0xFFFFu) << 16)); // dim0 hi | dim1 lo
    g1[3] = (int)((rowExtent >> 16) | ((unsigned)KCHUNK << 16));    // dim1 hi | tile_dim0
    g1[4] = 64;                                             // tile_dim1 = 64 rows
    g1[5] = (int)(unsigned)(strideElems & 0xFFFFFFFFu);     // dim0_stride lo
    g1[6] = (int)(unsigned)(strideElems >> 32);             // dim0_stride hi
    g1[7] = 0;
    i32x4 gz4 = {0, 0, 0, 0};
    i32x8 gz8 = {0, 0, 0, 0, 0, 0, 0, 0};
    __builtin_amdgcn_tensor_load_to_lds(g0, g1, gz4, gz4, gz8, 0);
}
#endif

// ------------------------------ prep kernels -------------------------------

// Tiled transpose f32[R x C] -> bf16[C x R]; grid.z batches.
__global__ __launch_bounds__(256)
void transpose_f32_to_bf16(const float* __restrict__ src, __bf16* __restrict__ dst,
                           int R, int Ccols, long sst, long dstst) {
    __shared__ float tl[32][33];
    const int z  = blockIdx.z;
    const int c0 = blockIdx.x * 32, r0 = blockIdx.y * 32;
    const int tx = threadIdx.x, ty = threadIdx.y;
    const float* s = src + (long)z * sst;
    __bf16*      d = dst + (long)z * dstst;
#pragma unroll
    for (int j = 0; j < 4; ++j) {
        int r = r0 + ty + j * 8;
        tl[ty + j * 8][tx] = s[(long)r * Ccols + c0 + tx];
    }
    __syncthreads();
#pragma unroll
    for (int j = 0; j < 4; ++j) {
        int cc = c0 + ty + j * 8;
        d[(long)cc * R + r0 + tx] = (__bf16)tl[tx][ty + j * 8];
    }
}

// RBF kernel matrix K[b,m,n] = exp(-(g_m - p_bn)^2 / (2 sigma^2)), bf16.
__global__ __launch_bounds__(256)
void rbf_k(const float* __restrict__ pos, const float* __restrict__ sigma,
           __bf16* __restrict__ Kb) {
    long idx = (long)blockIdx.x * blockDim.x + threadIdx.x;
    const long total = (long)BB * MM * NN;
    if (idx >= total) return;
    int  n = (int)(idx % NN);
    long r = idx / NN;
    int  m = (int)(r % MM);
    int  b = (int)(r / MM);
    float s   = sigma[0];
    float inv = 1.0f / (2.0f * s * s);
    float gm  = (float)m / (float)(MM - 1);
    float d   = gm - pos[(long)b * NN + n];
    Kb[idx] = (__bf16)expf(-d * d * inv);
}

// mix[b,m,h*DH+dh] = sum_k alpha[h,k] * field[b,m,k*DH+dh]  (bf16 out)
__global__ __launch_bounds__(256)
void headmix_k(const float* __restrict__ field, const float* __restrict__ alpha,
               __bf16* __restrict__ mixb) {
    __shared__ float sa[HH * HH];
    if (threadIdx.x < HH * HH) sa[threadIdx.x] = alpha[threadIdx.x];
    __syncthreads();
    long idx = (long)blockIdx.x * blockDim.x + threadIdx.x;
    const long total = (long)BB * MM * DD;
    if (idx >= total) return;
    int  d       = (int)(idx % DD);
    long rowbase = idx - d;
    int  h  = d / DHD;
    int  dh = d % DHD;
    float acc = 0.0f;
#pragma unroll
    for (int k = 0; k < HH; ++k)
        acc += sa[h * HH + k] * field[rowbase + (long)k * DHD + dh];
    mixb[idx] = (__bf16)acc;
}

// ------------------------------ WMMA GEMM ----------------------------------
// C[M x Ncols] = A[M x K](bf16,row-major) * Bt[Ncols x K](bf16,row-major)^T
// TDM double-buffers 64x64 A and B tiles in LDS; waves read fragments via
// ds_load_b128 and issue v_wmma_f32_16x16x32_bf16.
// mode 0: C = acc (+bias)        mode 1: C = res + acc + bias
// mode 2: outb = bf16(gelu(acc+bias))
__global__ __launch_bounds__(128)
void gemm_bf16_wmma(const __bf16* __restrict__ A, long lda, long sA,
                    const __bf16* __restrict__ Bt, long ldb, long sB,
                    float* __restrict__ C, const float* __restrict__ res,
                    const float* __restrict__ bias, __bf16* __restrict__ outb,
                    long ldc, long sC, int Kdim, int Mrows, int Ncols, int mode) {
    __shared__ __align__(16) char smem[4 * TILE_BYTES];  // A0,A1,B0,B1

    const int lane = threadIdx.x & 31;
    const int wave = threadIdx.x >> 5;
    const int z    = blockIdx.z;
    const int rowB = blockIdx.y * 64;                  // block row base
    const int col0 = blockIdx.x * 64;
    const int kselA = (lane >> 4) << 4;                // 0 or 16
    const int kselB = (lane >> 4) << 4;                // 0 or 16
    const int l15   = lane & 15;

    const __bf16* Ag = A  + (long)z * sA + (long)rowB * lda;  // 64-row panel
    const __bf16* Bg = Bt + (long)z * sB + (long)col0 * ldb;  // 64-col panel
    const unsigned smemBase = (unsigned)(unsigned long)(&smem[0]);

    v8f acc[4] = {};
    const int stages = Kdim / KCHUNK;

#if USE_TDM
    if (wave == 0) {
        tdm_load_tile(smemBase + 0 * TILE_BYTES, Ag, (unsigned)Kdim,
                      (unsigned)Mrows, (unsigned long)lda);
        tdm_load_tile(smemBase + 2 * TILE_BYTES, Bg, (unsigned)Kdim,
                      (unsigned)Ncols, (unsigned long)ldb);
    }
#endif

    for (int s = 0; s < stages; ++s) {
#if USE_TDM
        const int cur = s & 1;
        if (wave == 0) {
            if (s + 1 < stages) {
                const int nxt = cur ^ 1;
                tdm_load_tile(smemBase + nxt * TILE_BYTES,
                              Ag + (long)(s + 1) * KCHUNK, (unsigned)Kdim,
                              (unsigned)Mrows, (unsigned long)lda);
                tdm_load_tile(smemBase + (2 + nxt) * TILE_BYTES,
                              Bg + (long)(s + 1) * KCHUNK, (unsigned)Kdim,
                              (unsigned)Ncols, (unsigned long)ldb);
                // in-order completion: <=2 outstanding => stage s has landed
                __builtin_amdgcn_s_wait_tensorcnt(2);
            } else {
                __builtin_amdgcn_s_wait_tensorcnt(0);
            }
        }
        __syncthreads();
#else
        const int cur = 0;
        // cooperative fallback copy (same padded pitch)
        for (int idx = threadIdx.x; idx < 64 * 8; idx += 128) {
            int row = idx >> 3, seg = idx & 7;
            *(v8bf*)(smem + 0 * TILE_BYTES + ((long)row * TPITCH + seg * 8) * 2) =
                *(const v8bf*)(Ag + (long)row * lda + (long)s * KCHUNK + seg * 8);
            *(v8bf*)(smem + 2 * TILE_BYTES + ((long)row * TPITCH + seg * 8) * 2) =
                *(const v8bf*)(Bg + (long)row * ldb + (long)s * KCHUNK + seg * 8);
        }
        __syncthreads();
#endif
        const __bf16* La = (const __bf16*)(smem + cur * TILE_BYTES);
        const __bf16* Lb = (const __bf16*)(smem + (2 + cur) * TILE_BYTES);
        const __bf16* Lar = La + (wave * 16 + l15) * TPITCH;
#pragma unroll
        for (int kk = 0; kk < KCHUNK; kk += 32) {
            // hoist all fragment loads of this k-step so the ds_load_b128s
            // clause together and the 4 WMMAs issue back-to-back after one wait
            v16bf a  = load_a_frag(Lar + kk + kselA);
            v16bf b0 = load_b_frag(Lb + (0 * 16 + l15) * TPITCH + kk + kselB);
            v16bf b1 = load_b_frag(Lb + (1 * 16 + l15) * TPITCH + kk + kselB);
            v16bf b2 = load_b_frag(Lb + (2 * 16 + l15) * TPITCH + kk + kselB);
            v16bf b3 = load_b_frag(Lb + (3 * 16 + l15) * TPITCH + kk + kselB);
            acc[0] = __builtin_amdgcn_wmma_f32_16x16x32_bf16(
                false, a, false, b0, (short)0, acc[0], false, false);
            acc[1] = __builtin_amdgcn_wmma_f32_16x16x32_bf16(
                false, a, false, b1, (short)0, acc[1], false, false);
            acc[2] = __builtin_amdgcn_wmma_f32_16x16x32_bf16(
                false, a, false, b2, (short)0, acc[2], false, false);
            acc[3] = __builtin_amdgcn_wmma_f32_16x16x32_bf16(
                false, a, false, b3, (short)0, acc[3], false, false);
        }
        __syncthreads();
    }

    // ------------------------------ epilogue -------------------------------
    const int rbase = rowB + wave * 16 + ((lane >> 4) << 3);
    float*       Cb = C    ? C    + (long)z * sC : nullptr;
    const float* Rb = res  ? res  + (long)z * sC : nullptr;
    __bf16*      Ob = outb ? outb + (long)z * sC : nullptr;
#pragma unroll
    for (int t = 0; t < 4; ++t) {
        int   col = col0 + t * 16 + l15;
        float bi  = bias ? bias[col] : 0.0f;
#pragma unroll
        for (int r = 0; r < 8; ++r) {
            long  idx = (long)(rbase + r) * ldc + col;
            float v   = acc[t][r] + bi;
            if (mode == 1) v += Rb[idx];
            if (mode == 2) Ob[idx] = (__bf16)gelu_tanh(v);
            else           Cb[idx] = v;
        }
    }
}

// ------------------------------ diffusion ----------------------------------
__global__ __launch_bounds__(256)
void diffuse_k(float* __restrict__ f, __bf16* __restrict__ outb,
               const float* __restrict__ coef) {
    __shared__ float tile[MM * 16];
    const int b  = blockIdx.y;
    const int d0 = blockIdx.x * 16;
    const int dl = threadIdx.x & 15;
    const int mb = threadIdx.x >> 4;        // 0..15, each owns 32 rows
    const float cf = coef[d0 + dl] * DT_STEP;
    float* fb = f + (long)b * MM * DD;

#pragma unroll
    for (int i = 0; i < 32; ++i) {
        int m = mb + 16 * i;
        tile[m * 16 + dl] = fb[(long)m * DD + d0 + dl];
    }
    __syncthreads();

    for (int s = 0; s < PDE_STEPS; ++s) {
        float nv[32];
#pragma unroll
        for (int i = 0; i < 32; ++i) {
            int   m = mb + 16 * i;
            float c = tile[m * 16 + dl];
            float l = (m == 0)      ? c : tile[(m - 1) * 16 + dl];
            float r = (m == MM - 1) ? c : tile[(m + 1) * 16 + dl];
            nv[i] = c + cf * (l - 2.0f * c + r);
        }
        __syncthreads();
#pragma unroll
        for (int i = 0; i < 32; ++i) tile[(mb + 16 * i) * 16 + dl] = nv[i];
        __syncthreads();
    }

#pragma unroll
    for (int i = 0; i < 32; ++i) {
        int   m = mb + 16 * i;
        float v = tile[m * 16 + dl];
        long  gi = (long)b * MM * DD + (long)m * DD + d0 + dl;
        f[gi] = v;
        if (outb) outb[gi] = (__bf16)v;
    }
}

// ---------------------- sampling + residual + LayerNorm --------------------
__global__ __launch_bounds__(256)
void sample_ln_k(const float* __restrict__ field, const float* __restrict__ emb,
                 const float* __restrict__ pos, const float* __restrict__ g,
                 const float* __restrict__ beta, float* __restrict__ enh32,
                 __bf16* __restrict__ enhb) {
    __shared__ float red[256];
    const int  row = blockIdx.x;            // b*N + n
    const int  b   = row / NN;
    const int  t   = threadIdx.x;
    const float* fb = field + (long)b * MM * DD;

    float p  = fminf(fmaxf(pos[row], 0.0f), 1.0f);
    float u  = p * (float)(MM - 1);
    int   i0 = (int)floorf(u);
    i0 = (i0 < 0) ? 0 : ((i0 > MM - 2) ? MM - 2 : i0);
    float w = u - (float)i0;

    float x[2];
#pragma unroll
    for (int j = 0; j < 2; ++j) {
        int   d  = t + j * 256;
        float f0 = fb[(long)i0 * DD + d];
        float f1 = fb[(long)(i0 + 1) * DD + d];
        x[j] = (1.0f - w) * f0 + w * f1 + emb[(long)row * DD + d];
    }

    red[t] = x[0] + x[1];
    __syncthreads();
    for (int o = 128; o > 0; o >>= 1) { if (t < o) red[t] += red[t + o]; __syncthreads(); }
    float mu = red[0] / (float)DD;
    __syncthreads();
    float v0 = x[0] - mu, v1 = x[1] - mu;
    red[t] = v0 * v0 + v1 * v1;
    __syncthreads();
    for (int o = 128; o > 0; o >>= 1) { if (t < o) red[t] += red[t + o]; __syncthreads(); }
    float inv = rsqrtf(red[0] / (float)DD + 1e-5f);

#pragma unroll
    for (int j = 0; j < 2; ++j) {
        int   d = t + j * 256;
        float y = (x[j] - mu) * inv * g[d] + beta[d];
        enh32[(long)row * DD + d] = y;
        enhb[(long)row * DD + d]  = (__bf16)y;
    }
}

// In-place row LayerNorm over D=512.
__global__ __launch_bounds__(256)
void ln_k(float* __restrict__ y, const float* __restrict__ g,
          const float* __restrict__ beta) {
    __shared__ float red[256];
    const int row = blockIdx.x;
    const int t   = threadIdx.x;
    float x[2];
#pragma unroll
    for (int j = 0; j < 2; ++j) x[j] = y[(long)row * DD + t + j * 256];
    red[t] = x[0] + x[1];
    __syncthreads();
    for (int o = 128; o > 0; o >>= 1) { if (t < o) red[t] += red[t + o]; __syncthreads(); }
    float mu = red[0] / (float)DD;
    __syncthreads();
    float v0 = x[0] - mu, v1 = x[1] - mu;
    red[t] = v0 * v0 + v1 * v1;
    __syncthreads();
    for (int o = 128; o > 0; o >>= 1) { if (t < o) red[t] += red[t + o]; __syncthreads(); }
    float inv = rsqrtf(red[0] / (float)DD + 1e-5f);
#pragma unroll
    for (int j = 0; j < 2; ++j) {
        int d = t + j * 256;
        y[(long)row * DD + d] = (x[j] - mu) * inv * g[d] + beta[d];
    }
}

// ------------------------------ orchestration ------------------------------

extern "C" void kernel_launch(void* const* d_in, const int* in_sizes, int n_in,
                              void* d_out, int out_size, void* d_ws, size_t ws_size,
                              hipStream_t stream) {
    const float* emb    = (const float*)d_in[0];
    const float* pos    = (const float*)d_in[1];
    const float* sigma  = (const float*)d_in[2];
    const float* alpha  = (const float*)d_in[3];
    const float* w_int  = (const float*)d_in[4];
    const float* b_int  = (const float*)d_in[5];
    const float* dcoef  = (const float*)d_in[6];
    const float* W1     = (const float*)d_in[7];
    const float* b1     = (const float*)d_in[8];
    const float* W2     = (const float*)d_in[9];
    const float* b2     = (const float*)d_in[10];
    const float* ecoef  = (const float*)d_in[11];
    const float* ln1_g  = (const float*)d_in[12];
    const float* ln1_b  = (const float*)d_in[13];
    const float* Wout   = (const float*)d_in[14];
    const float* bout   = (const float*)d_in[15];
    const float* ln2_g  = (const float*)d_in[16];
    const float* ln2_b  = (const float*)d_in[17];
    float* out = (float*)d_out;

    // ---- workspace layout (all multiples of 256B); ~145 MB total ----
    char* w = (char*)d_ws;
    size_t off = 0;
    auto take = [&](size_t bytes) { char* p = w + off; off += (bytes + 255) & ~(size_t)255; return p; };
    __bf16* embT    = (__bf16*)take((size_t)BB * DD * NN * 2);   // 33.5 MB
    __bf16* Kb      = (__bf16*)take((size_t)BB * MM * NN * 2);   // 33.5 MB
    float*  enh32   = (float*)w;  // aliases embT+Kb (dead after field GEMM): 67 MB
    float*  field32 = (float*)take((size_t)BB * MM * DD * 4);    // 16.8 MB
    __bf16* fieldb  = (__bf16*)take((size_t)BB * MM * DD * 2);   // 8.4 MB
    __bf16* mixb    = (__bf16*)take((size_t)BB * MM * DD * 2);   // 8.4 MB
    __bf16* hidb    = (__bf16*)take((size_t)BB * MM * DD * 2);   // 8.4 MB
    __bf16* enhb    = (__bf16*)take((size_t)BB * NN * DD * 2);   // 33.5 MB
    __bf16* wT0     = (__bf16*)take((size_t)DD * DD * 2);
    __bf16* wT1     = (__bf16*)take((size_t)DD * DD * 2);
    __bf16* wT2     = (__bf16*)take((size_t)DD * DD * 2);
    __bf16* wT3     = (__bf16*)take((size_t)DD * DD * 2);
    (void)ws_size; (void)n_in; (void)in_sizes; (void)out_size;

    dim3 tb(32, 8, 1);
    // weights -> bf16, transposed to [N][K] for B-fragment loads
    transpose_f32_to_bf16<<<dim3(16, 16, 1), tb, 0, stream>>>(w_int, wT0, DD, DD, 0, 0);
    transpose_f32_to_bf16<<<dim3(16, 16, 1), tb, 0, stream>>>(W1,    wT1, DD, DD, 0, 0);
    transpose_f32_to_bf16<<<dim3(16, 16, 1), tb, 0, stream>>>(W2,    wT2, DD, DD, 0, 0);
    transpose_f32_to_bf16<<<dim3(16, 16, 1), tb, 0, stream>>>(Wout,  wT3, DD, DD, 0, 0);
    // embeddings [B,N,D] -> embT [B,D,N] bf16
    transpose_f32_to_bf16<<<dim3(DD / 32, NN / 32, BB), tb, 0, stream>>>(
        emb, embT, NN, DD, (long)NN * DD, (long)NN * DD);
    // RBF kernel matrix, bf16
    rbf_k<<<dim3((unsigned)((long)BB * MM * NN / 256)), 256, 0, stream>>>(pos, sigma, Kb);

    // 1) field = K @ emb : per-batch [512 x 2048] * [2048 x 512]
    gemm_bf16_wmma<<<dim3(DD / 64, MM / 64, BB), 128, 0, stream>>>(
        Kb, NN, (long)MM * NN, embT, NN, (long)DD * NN,
        field32, nullptr, nullptr, nullptr, DD, (long)MM * DD, NN, MM, DD, 0);

    // 2) head coupling + projection: field += mix @ w_int + b_int
    headmix_k<<<dim3((unsigned)((long)BB * MM * DD / 256)), 256, 0, stream>>>(field32, alpha, mixb);
    gemm_bf16_wmma<<<dim3(DD / 64, (BB * MM) / 64, 1), 128, 0, stream>>>(
        mixb, DD, 0, wT0, DD, 0,
        field32, field32, b_int, nullptr, DD, 0, DD, BB * MM, DD, 1);

    // 3) diffusion (also emits bf16 copy feeding the MLP GEMM)
    diffuse_k<<<dim3(DD / 16, BB, 1), 256, 0, stream>>>(field32, fieldb, dcoef);

    // 4) field += gelu(field@W1 + b1) @ W2 + b2
    gemm_bf16_wmma<<<dim3(DD / 64, (BB * MM) / 64, 1), 128, 0, stream>>>(
        fieldb, DD, 0, wT1, DD, 0,
        nullptr, nullptr, b1, hidb, DD, 0, DD, BB * MM, DD, 2);
    gemm_bf16_wmma<<<dim3(DD / 64, (BB * MM) / 64, 1), 128, 0, stream>>>(
        hidb, DD, 0, wT2, DD, 0,
        field32, field32, b2, nullptr, DD, 0, DD, BB * MM, DD, 1);

    // 5) evolution diffusion (f32 in place)
    diffuse_k<<<dim3(DD / 16, BB, 1), 256, 0, stream>>>(field32, nullptr, ecoef);

    // 6+7a) sample at positions, +emb residual, LN1 -> enh (f32 + bf16)
    sample_ln_k<<<dim3(BB * NN), 256, 0, stream>>>(field32, emb, pos, ln1_g, ln1_b, enh32, enhb);

    // 7b) out = enh + enh@Wout + bout  (into d_out)
    gemm_bf16_wmma<<<dim3(DD / 64, (BB * NN) / 64, 1), 128, 0, stream>>>(
        enhb, DD, 0, wT3, DD, 0,
        out, enh32, bout, nullptr, DD, 0, DD, BB * NN, DD, 1);

    // 7c) LN2 in place on d_out
    ln_k<<<dim3(BB * NN), 256, 0, stream>>>(out, ln2_g, ln2_b);
}
